// MaskRCNN_40372692583125
// MI455X (gfx1250) — matrix operand order repaired and checked by reading, not compiled, not run on hardware
//
#include <hip/hip_runtime.h>
#include <hip/hip_bf16.h>

// ---------------------------------------------------------------------------
// RPN head for MI455X (gfx1250): bf16 WMMA implicit-GEMM conv3x3 with
// async-to-LDS double-buffered weight staging + fused cls/reg 1x1 GEMM,
// anchor decode, bitonic top-k, LDS greedy NMS, gather.
// ---------------------------------------------------------------------------

typedef __attribute__((ext_vector_type(16))) __bf16 v16bf;
typedef __attribute__((ext_vector_type(8)))  float  v8f;

#define BATCH   4
#define CIN     512
#define FH      64
#define M_TOTAL (BATCH * FH * FH)     // 16384 output pixels
#define NANCH   49152                 // 64*64*12 anchors per image
#define NPRE    12000
#define NPOST   2000
#define NSORT   65536
#define NMS_TH  0.7f

#if __has_builtin(__builtin_amdgcn_global_load_async_to_lds_b128)
#define HAVE_ASYNC_LDS 1
// Builtin signature (from hipcc diagnostic): param0 is
//   'int __attribute__((vector_size(16))) __device__ *'  (AS1)
// param1 is the LDS-side pointer (AS3), then imm offset + imm cpol.
typedef int v4i_vs __attribute__((vector_size(16)));
typedef __attribute__((address_space(1))) v4i_vs as1_v4i;
typedef __attribute__((address_space(3))) v4i_vs as3_v4i;
#else
#define HAVE_ASYNC_LDS 0
#endif

static __device__ __forceinline__ unsigned short f2bf(float f) {
  unsigned u = __builtin_bit_cast(unsigned, f);
  u = (u + 0x7FFFu + ((u >> 16) & 1u)) >> 16;   // round-to-nearest-even
  return (unsigned short)u;
}

union FragU { uint4 q[2]; v16bf v; };

static __device__ __forceinline__ v16bf load_frag2(const unsigned short* p0,
                                                   const unsigned short* p1) {
  FragU u;
  u.q[0] = *(const uint4*)(const void*)p0;   // 16B-aligned by construction
  u.q[1] = *(const uint4*)(const void*)p1;
  return u.v;
}

// 16B global -> LDS copy: async path on gfx1250 (ASYNCcnt tracked),
// plain load+ds_store fallback otherwise.
static __device__ __forceinline__ void lds_copy16(const unsigned short* g,
                                                  unsigned char* l) {
#if HAVE_ASYNC_LDS
  __builtin_amdgcn_global_load_async_to_lds_b128(
      (as1_v4i*)(unsigned long long)g,
      (as3_v4i*)(unsigned)(unsigned long long)l, 0, 0);
#else
  *(uint4*)(void*)l = *(const uint4*)(const void*)g;
#endif
}

static __device__ __forceinline__ void wait_async0() {
#if HAVE_ASYNC_LDS
#if __has_builtin(__builtin_amdgcn_s_wait_asynccnt)
  __builtin_amdgcn_s_wait_asynccnt(0);
#else
  asm volatile("s_wait_asynccnt 0x0" ::: "memory");
#endif
#endif
}

// ---------------------------------------------------------------------------
// Packing kernels
// ---------------------------------------------------------------------------

// fp32 NCHW -> bf16 NHWC with 1-pixel zero pad: [B][66][66][512]
__global__ void pad_feat(const float* __restrict__ f,
                         unsigned short* __restrict__ out, int total) {
  int idx = blockIdx.x * blockDim.x + threadIdx.x;
  if (idx >= total) return;
  int c  = idx & 511;
  int t  = idx >> 9;
  int xx = t % 66; t /= 66;
  int yy = t % 66;
  int b  = t / 66;
  int x = xx - 1, y = yy - 1;
  float v = 0.f;
  if ((unsigned)x < 64u && (unsigned)y < 64u)
    v = f[(((size_t)b * 512 + c) * 64 + y) * 64 + x];
  out[idx] = f2bf(v);
}

// OIHW fp32 -> WMMA-B-fragment-swizzled bf16: [32 ntile][144 kchunk][32 lane][16]
__global__ void pack_w3(const float* __restrict__ w,
                        unsigned short* __restrict__ out, int total) {
  int idx = blockIdx.x * blockDim.x + threadIdx.x;
  if (idx >= total) return;
  int e     = idx & 15;
  int lane  = (idx >> 4) & 31;
  int chunk = (idx >> 9) % 144;
  int tile  = (idx >> 9) / 144;
  int n = tile * 16 + (lane & 15);
  int K = chunk * 32 + 8 * (lane >> 4) + (e & 7) + 16 * (e >> 3);
  int posK = K >> 9, c = K & 511;
  int dy = posK / 3, dx = posK % 3;
  float v = w[(((size_t)n * 512 + c) * 3 + dy) * 3 + dx];
  out[idx] = f2bf(v);
}

// cls(24) + reg(48) fused, padded to N=80: [5 ntile][16 kchunk][32 lane][16]
__global__ void pack_wh(const float* __restrict__ wc, const float* __restrict__ wr,
                        unsigned short* __restrict__ out, int total) {
  int idx = blockIdx.x * blockDim.x + threadIdx.x;
  if (idx >= total) return;
  int e    = idx & 15;
  int lane = (idx >> 4) & 31;
  int cc   = (idx >> 9) % 16;
  int tile = (idx >> 9) / 16;
  int n = tile * 16 + (lane & 15);
  int k = cc * 32 + 8 * (lane >> 4) + (e & 7) + 16 * (e >> 3);
  float v = 0.f;
  if (n < 24)      v = wc[(size_t)n * 512 + k];
  else if (n < 72) v = wr[(size_t)(n - 24) * 512 + k];
  out[idx] = f2bf(v);
}

// ---------------------------------------------------------------------------
// conv3x3 512->512 as implicit GEMM: M=16384, N=512, K=4608.
// 8-wave workgroup computes 128(M) x 64(N). Per K-chunk (32 K), the 4KB
// B slice is staged into LDS (double-buffered) via async-to-LDS copies and
// shared by all 8 waves; A fragments are two direct b128 loads per lane,
// software-pipelined one chunk ahead. fp32 accumulate, +bias, ReLU, bf16 out.
// ---------------------------------------------------------------------------
__global__ __launch_bounds__(256) void conv3_wmma(
    const unsigned short* __restrict__ feat,   // [B][66][66][512] bf16
    const unsigned short* __restrict__ w3p,    // [32][144][32][16] bf16
    const float* __restrict__ bias,            // [512]
    unsigned short* __restrict__ hout)         // [16384][512] bf16
{
  __shared__ unsigned char ldsB[2 * 4096];     // double-buffered B chunk

  const int tid  = threadIdx.x;
  const int lane = tid & 31;
  const int wv   = tid >> 5;                   // wave id 0..7
  const int mt   = blockIdx.x * 8 + wv;        // 16-row M tile
  const int nt   = blockIdx.y;                 // 64-col N tile
  const int m    = mt * 16 + (lane & 15);
  const int hh   = lane >> 4;
  const int b = m >> 12, y = (m >> 6) & 63, x = m & 63;
  const unsigned short* base0 = feat + ((size_t)((b * 66 + y) * 66 + x)) * 512;

  // B staging: thread tid copies 16B; 256 threads cover the 4KB chunk slice.
  const int tile_s = tid >> 6;                 // 0..3 (n sub-tile)
  const int r      = tid & 63;                 // 64 x 16B = 1KB per sub-tile
  const unsigned short* bsrc_base =
      w3p + ((size_t)(nt * 4 + tile_s)) * 144 * 512 + r * 8;
  unsigned char* bdst0 = ldsB + tile_s * 1024 + r * 16;

  auto issueB = [&](int chunk, int buf) {
    lds_copy16(bsrc_base + (size_t)chunk * 512, bdst0 + buf * 4096);
  };
  auto loadA = [&](int chunk) -> FragU {
    int pos = chunk >> 4, cc = chunk & 15;
    int dy = (pos >= 6) ? 2 : ((pos >= 3) ? 1 : 0);
    int dx = pos - dy * 3;
    const unsigned short* ar = base0 + (dy * 66 + dx) * 512 + cc * 32 + 8 * hh;
    FragU u;
    u.q[0] = *(const uint4*)(const void*)ar;
    u.q[1] = *(const uint4*)(const void*)(ar + 16);
    return u;
  };
  auto ldsfrag = [&](const unsigned char* p) -> v16bf {
    FragU u;
    u.q[0] = *(const uint4*)(const void*)p;
    u.q[1] = *(const uint4*)(const void*)(p + 16);
    return u.v;
  };

  v8f acc0 = {}, acc1 = {}, acc2 = {}, acc3 = {};

  issueB(0, 0);
  FragU Acur = loadA(0);
  wait_async0();
  __syncthreads();

  for (int chunk = 0; chunk < 144; ++chunk) {
    const int buf = chunk & 1;
    const bool more = (chunk + 1) < 144;
    if (more) issueB(chunk + 1, buf ^ 1);      // fill other buffer
    FragU Anxt;
    if (more) Anxt = loadA(chunk + 1);         // pipeline A one chunk ahead

    const unsigned char* lb = ldsB + buf * 4096 + lane * 32;
    v16bf B0 = ldsfrag(lb + 0 * 1024);
    v16bf B1 = ldsfrag(lb + 1 * 1024);
    v16bf B2 = ldsfrag(lb + 2 * 1024);
    v16bf B3 = ldsfrag(lb + 3 * 1024);

    acc0 = __builtin_amdgcn_wmma_f32_16x16x32_bf16(false, Acur.v, false, B0,
                                                   (short)0, acc0, false, false);
    acc1 = __builtin_amdgcn_wmma_f32_16x16x32_bf16(false, Acur.v, false, B1,
                                                   (short)0, acc1, false, false);
    acc2 = __builtin_amdgcn_wmma_f32_16x16x32_bf16(false, Acur.v, false, B2,
                                                   (short)0, acc2, false, false);
    acc3 = __builtin_amdgcn_wmma_f32_16x16x32_bf16(false, Acur.v, false, B3,
                                                   (short)0, acc3, false, false);
    Acur = Anxt;
    if (more) {
      wait_async0();                           // this wave's copies landed
      __syncthreads();                         // everyone's copies landed
    }
  }

  // C layout: lanes 0-15 rows M=v, lanes 16-31 rows M=v+8, col N = lane%16
  #pragma unroll
  for (int v = 0; v < 8; ++v) {
    const size_t row = (size_t)(mt * 16 + 8 * hh + v) * 512;
    {
      int n = nt * 64 + 0 * 16 + (lane & 15);
      float val = acc0[v] + bias[n];
      hout[row + n] = f2bf(val > 0.f ? val : 0.f);
    }
    {
      int n = nt * 64 + 1 * 16 + (lane & 15);
      float val = acc1[v] + bias[n];
      hout[row + n] = f2bf(val > 0.f ? val : 0.f);
    }
    {
      int n = nt * 64 + 2 * 16 + (lane & 15);
      float val = acc2[v] + bias[n];
      hout[row + n] = f2bf(val > 0.f ? val : 0.f);
    }
    {
      int n = nt * 64 + 3 * 16 + (lane & 15);
      float val = acc3[v] + bias[n];
      hout[row + n] = f2bf(val > 0.f ? val : 0.f);
    }
  }
}

// ---------------------------------------------------------------------------
// Fused cls+reg 1x1 convs: GEMM M=16384, K=512, N=80 (72 used), WMMA.
// Scatters into (B,49152,2) score and (B,49152,4) offset layouts.
// ---------------------------------------------------------------------------
__global__ __launch_bounds__(32) void head_wmma(
    const unsigned short* __restrict__ hbuf,   // [16384][512] bf16
    const unsigned short* __restrict__ whp,    // [5][16][32][16] bf16
    const float* __restrict__ b_cls, const float* __restrict__ b_reg,
    float* __restrict__ score,                 // [4][49152][2]
    float* __restrict__ offs)                  // [4][49152][4]
{
  const int lane = threadIdx.x;
  const int mt = blockIdx.x, nt = blockIdx.y;
  const int m  = mt * 16 + (lane & 15);
  const int hh = lane >> 4;
  const unsigned short* arow = hbuf + (size_t)m * 512;

  v8f acc = {};
  #pragma unroll 4
  for (int cc = 0; cc < 16; ++cc) {
    const int c0 = cc * 32;
    v16bf A = load_frag2(arow + c0 + 8 * hh, arow + c0 + 16 + 8 * hh);
    const unsigned short* wb = whp + (((size_t)nt * 16 + cc) * 32 + lane) * 16;
    v16bf B = load_frag2(wb, wb + 8);
    acc = __builtin_amdgcn_wmma_f32_16x16x32_bf16(false, A, false, B,
                                                  (short)0, acc, false, false);
  }

  const int ch = nt * 16 + (lane & 15);
  float bia = 0.f;
  if (ch < 24)      bia = b_cls[ch];
  else if (ch < 72) bia = b_reg[ch - 24];

  #pragma unroll
  for (int v = 0; v < 8; ++v) {
    const int mr = mt * 16 + 8 * hh + v;
    const int b = mr >> 12, p = mr & 4095;
    float val = acc[v] + bia;
    if (ch < 24) {
      int a = ch >> 1, s = ch & 1;
      score[((size_t)b * NANCH + (size_t)p * 12 + a) * 2 + s] = val;
    } else if (ch < 72) {
      int t = ch - 24;
      offs[((size_t)b * NANCH + (size_t)p * 12 + (t >> 2)) * 4 + (t & 3)] = val;
    }
  }
}

// ---------------------------------------------------------------------------
// Anchor generation + offset2bbox + clip
// ---------------------------------------------------------------------------
__global__ void decode_anchors(const float* __restrict__ offs,
                               float* __restrict__ bbox, int total) {
  int idx = blockIdx.x * blockDim.x + threadIdx.x;
  if (idx >= total) return;                      // total = 4*49152
  int A = idx % NANCH;
  int p = A / 12, a = A % 12;
  int i = p >> 6, j = p & 63;
  const float areas[4]  = {2000.f, 1000.f, 2000.f / 9.f, 30.f};
  const float ratios[3] = {0.5f, 1.f, 2.f};
  float area = areas[a & 3], r = ratios[a >> 2];
  float ah = sqrtf(area / r), aw = sqrtf(area * r);
  float cy = 7.5f + 16.f * (float)i;
  float cx = 7.5f + 16.f * (float)j;
  float hhv = 2.f * ah, wwv = 2.f * aw;
  const float* o = offs + (size_t)idx * 4;
  float dy = o[0], dx = o[1];
  float dh = 0.5f * expf(o[2]);
  float dw = 0.5f * expf(o[3]);
  float y1 = fminf(fmaxf(cy + hhv * (dy - dh), 0.f), 1024.f);
  float x1 = fminf(fmaxf(cx + wwv * (dx - dw), 0.f), 1024.f);
  float y2 = fminf(fmaxf(cy + hhv * (dy + dh), 0.f), 1024.f);
  float x2 = fminf(fmaxf(cx + wwv * (dx + dw), 0.f), 1024.f);
  float* bb = bbox + (size_t)idx * 4;
  bb[0] = y1; bb[1] = x1; bb[2] = y2; bb[3] = x2;
}

// ---------------------------------------------------------------------------
// Exact descending top-k via full bitonic sort of packed u64 keys.
// key = (~orderable(score)) << 32 | idx  -> ascending sort == desc score,
// ties broken by lower original index (matches jax.lax.top_k).
// ---------------------------------------------------------------------------
__global__ void init_keys(const float* __restrict__ score,
                          unsigned long long* __restrict__ keys, int total) {
  int idx = blockIdx.x * blockDim.x + threadIdx.x;
  if (idx >= total) return;                      // total = 4*65536
  int b = idx / NSORT, t = idx % NSORT;
  unsigned long long k;
  if (t < NANCH) {
    float s = score[((size_t)b * NANCH + t) * 2 + 1];
    unsigned u = __builtin_bit_cast(unsigned, s);
    u = (u & 0x80000000u) ? ~u : (u | 0x80000000u);
    k = ((unsigned long long)(~u) << 32) | (unsigned)t;
  } else {
    k = ~0ull;
  }
  keys[idx] = k;
}

__global__ void bitonic_pass(unsigned long long* __restrict__ keys, int k, int j) {
  int b = blockIdx.y;
  unsigned long long* K = keys + (size_t)b * NSORT;
  int i = blockIdx.x * blockDim.x + threadIdx.x;
  int ixj = i ^ j;
  if (ixj > i) {
    unsigned long long a = K[i], c = K[ixj];
    bool up = ((i & k) == 0);
    if ((a > c) == up) { K[i] = c; K[ixj] = a; }
  }
}

__global__ void stage_sorted(const unsigned long long* __restrict__ keys,
                             const float* __restrict__ bbox,
                             float* __restrict__ bbox_s, float* __restrict__ area_s,
                             int* __restrict__ order_idx, int total) {
  int idx = blockIdx.x * blockDim.x + threadIdx.x;
  if (idx >= total) return;                      // total = 4*12000
  int b = idx / NPRE, pos = idx % NPRE;
  unsigned orig = (unsigned)(keys[(size_t)b * NSORT + pos] & 0xFFFFFFFFull);
  order_idx[idx] = (int)orig;
  const float* bb = bbox + ((size_t)b * NANCH + orig) * 4;
  float y1 = bb[0], x1 = bb[1], y2 = bb[2], x2 = bb[3];
  float* d = bbox_s + (size_t)idx * 4;
  d[0] = y1; d[1] = x1; d[2] = y2; d[3] = x2;
  area_s[idx] = (y2 - y1) * (x2 - x1);
}

// ---------------------------------------------------------------------------
// Greedy NMS: one 1024-thread workgroup per image, suppressed flags in LDS.
// Suppress when !(iou < 0.7) to replicate jax NaN semantics exactly.
// ---------------------------------------------------------------------------
__global__ __launch_bounds__(1024) void nms_kernel(
    const float* __restrict__ bbox_s, const float* __restrict__ area_s,
    int* __restrict__ kept, int* __restrict__ counts) {
  __shared__ unsigned char sup[NPRE];
  const int b = blockIdx.x, tid = threadIdx.x;
  const float* BB = bbox_s + (size_t)b * NPRE * 4;
  const float* AR = area_s + (size_t)b * NPRE;
  for (int t = tid; t < NPRE; t += 1024) sup[t] = 0;
  __syncthreads();

  for (int i = 0; i < NPRE; ++i) {
    if (!sup[i]) {
      float iy1 = BB[i * 4 + 0], ix1 = BB[i * 4 + 1];
      float iy2 = BB[i * 4 + 2], ix2 = BB[i * 4 + 3];
      float ia = AR[i];
      for (int j = i + 1 + tid; j < NPRE; j += 1024) {
        if (sup[j]) continue;
        float ty = fmaxf(iy1, BB[j * 4 + 0]);
        float tx = fmaxf(ix1, BB[j * 4 + 1]);
        float by = fminf(iy2, BB[j * 4 + 2]);
        float bx = fminf(ix2, BB[j * 4 + 3]);
        float inter = (ty < by && tx < bx) ? (by - ty) * (bx - tx) : 0.f;
        float iou = inter / (ia + AR[j] - inter);
        if (!(iou < NMS_TH)) sup[j] = 1;
      }
    }
    __syncthreads();
  }

  if (tid == 0) {
    int cnt = 0;
    for (int pos = 0; pos < NPRE; ++pos) {
      if (!sup[pos]) {
        if (cnt < NPOST) kept[b * NPOST + cnt] = pos;
        cnt++;
      }
    }
    counts[b] = cnt < NPOST ? cnt : NPOST;
  }
}

// ---------------------------------------------------------------------------
// Final gather. Replicates reference exactly, including offset[safe] indexing
// the UNSORTED offset array with the sorted position.
// ---------------------------------------------------------------------------
__global__ void gather_out(const float* __restrict__ offs,
                           const float* __restrict__ score,
                           const float* __restrict__ bbox,
                           const int* __restrict__ kept,
                           const int* __restrict__ counts,
                           const int* __restrict__ order_idx,
                           float* __restrict__ out, int total) {
  int idx = blockIdx.x * blockDim.x + threadIdx.x;
  if (idx >= total) return;                      // total = 4*2000
  int b = idx / NPOST, t = idx % NPOST;
  float* o = out + (size_t)idx * 10;
  if (t < counts[b]) {
    int pos  = kept[b * NPOST + t];
    int orig = order_idx[b * NPRE + pos];
    const float* of = offs + ((size_t)b * NANCH + pos) * 4;   // offset[safe]!
    o[0] = of[0]; o[1] = of[1]; o[2] = of[2]; o[3] = of[3];
    const float* sc = score + ((size_t)b * NANCH + orig) * 2;
    o[4] = sc[0]; o[5] = sc[1];
    const float* bb = bbox + ((size_t)b * NANCH + orig) * 4;
    o[6] = bb[0]; o[7] = bb[1]; o[8] = bb[2]; o[9] = bb[3];
  } else {
    #pragma unroll
    for (int k = 0; k < 10; ++k) o[k] = 0.f;
  }
}

// ---------------------------------------------------------------------------
extern "C" void kernel_launch(void* const* d_in, const int* in_sizes, int n_in,
                              void* d_out, int out_size, void* d_ws, size_t ws_size,
                              hipStream_t stream) {
  (void)in_sizes; (void)n_in; (void)out_size; (void)ws_size;
  const float* features = (const float*)d_in[0];
  const float* w_inter  = (const float*)d_in[1];
  const float* b_inter  = (const float*)d_in[2];
  const float* w_cls    = (const float*)d_in[3];
  const float* b_cls    = (const float*)d_in[4];
  const float* w_reg    = (const float*)d_in[5];
  const float* b_reg    = (const float*)d_in[6];

  char* ws = (char*)d_ws;
  size_t off = 0;
  auto alloc = [&](size_t bytes) -> char* {
    char* p = ws + off;
    off = (off + bytes + 255) & ~(size_t)255;
    return p;
  };
  unsigned short* feat_pad = (unsigned short*)alloc((size_t)4 * 66 * 66 * 512 * 2);
  unsigned short* w3p      = (unsigned short*)alloc((size_t)32 * 144 * 32 * 16 * 2);
  unsigned short* whp      = (unsigned short*)alloc((size_t)5 * 16 * 32 * 16 * 2);
  unsigned short* hbuf     = (unsigned short*)alloc((size_t)M_TOTAL * 512 * 2);
  float* score             = (float*)alloc((size_t)4 * NANCH * 2 * 4);
  float* offs              = (float*)alloc((size_t)4 * NANCH * 4 * 4);
  float* bbox              = (float*)alloc((size_t)4 * NANCH * 4 * 4);
  unsigned long long* keys = (unsigned long long*)alloc((size_t)4 * NSORT * 8);
  float* bbox_s            = (float*)alloc((size_t)4 * NPRE * 4 * 4);
  float* area_s            = (float*)alloc((size_t)4 * NPRE * 4);
  int* order_idx           = (int*)alloc((size_t)4 * NPRE * 4);
  int* kept                = (int*)alloc((size_t)4 * NPOST * 4);
  int* counts              = (int*)alloc(64);

  { int tot = 4 * 66 * 66 * 512;
    pad_feat<<<(tot + 255) / 256, 256, 0, stream>>>(features, feat_pad, tot); }
  { int tot = 32 * 144 * 32 * 16;
    pack_w3<<<(tot + 255) / 256, 256, 0, stream>>>(w_inter, w3p, tot); }
  { int tot = 5 * 16 * 32 * 16;
    pack_wh<<<(tot + 255) / 256, 256, 0, stream>>>(w_cls, w_reg, whp, tot); }

  conv3_wmma<<<dim3(M_TOTAL / 128, 8), 256, 0, stream>>>(feat_pad, w3p, b_inter,
                                                         hbuf);
  head_wmma<<<dim3(M_TOTAL / 16, 5), 32, 0, stream>>>(hbuf, whp, b_cls, b_reg,
                                                      score, offs);

  { int tot = 4 * NANCH;
    decode_anchors<<<(tot + 255) / 256, 256, 0, stream>>>(offs, bbox, tot); }
  { int tot = 4 * NSORT;
    init_keys<<<(tot + 255) / 256, 256, 0, stream>>>(score, keys, tot); }
  for (int k = 2; k <= NSORT; k <<= 1)
    for (int j = k >> 1; j > 0; j >>= 1)
      bitonic_pass<<<dim3(NSORT / 256, 4), 256, 0, stream>>>(keys, k, j);
  { int tot = 4 * NPRE;
    stage_sorted<<<(tot + 255) / 256, 256, 0, stream>>>(keys, bbox, bbox_s,
                                                        area_s, order_idx, tot); }
  nms_kernel<<<4, 1024, 0, stream>>>(bbox_s, area_s, kept, counts);
  { int tot = 4 * NPOST;
    gather_out<<<(tot + 255) / 256, 256, 0, stream>>>(offs, score, bbox, kept,
                                                      counts, order_idx,
                                                      (float*)d_out, tot); }
}
